// LSTMPredictor_21440476742095
// MI455X (gfx1250) — matrix operand order repaired and checked by reading, not compiled, not run on hardware
//
#include <hip/hip_runtime.h>
#include <hip/hip_bf16.h>

// ---------------------------------------------------------------------------
// Autoregressive LSTM predictor for MI455X (gfx1250, wave32, WMMA).
// 137 GFLOP of (8192x256)@(256x1024) bf16 GEMM over 32 recurrent steps.
// Weights (~0.6MB bf16) are L2/WGP$-resident; h/c/x live in LDS.
// Round-3 fix: manual ping-pong B double-buffer in the K loop -> removes the
// 32x v_dual_mov_b32 register-copy chain the rolled single-buffer produced.
// ---------------------------------------------------------------------------

typedef __attribute__((ext_vector_type(16))) __bf16 v16bf;
typedef __attribute__((ext_vector_type(8)))  float  v8f;

union V16U { v16bf v; uint4 q[2]; };

#if defined(__has_builtin)
#if __has_builtin(__builtin_amdgcn_sched_barrier)
#define SCHED_FENCE() __builtin_amdgcn_sched_barrier(0)
#endif
#endif
#ifndef SCHED_FENCE
#define SCHED_FENCE()
#endif

// A-operand (16x32 bf16): lanes 0-15 hold K=0..15 of row M=lane,
// lanes 16-31 hold K=16..31 of row M=lane-16.  Two 16B loads per lane.
__device__ __forceinline__ v16bf load_a16(const unsigned short* p, int sel) {
  V16U t;
  const unsigned short* pp = p + sel * 16;
  t.q[0] = *(const uint4*)(pp);
  t.q[1] = *(const uint4*)(pp + 8);
  return t.v;
}

// B-operand (32x16 bf16): lane holds 16 contiguous K values of column N=lane%16
// (K=0..15 for lanes 0-15, K=16..31 for lanes 16-31).  One 32B load per lane.
__device__ __forceinline__ v16bf load_b16(const unsigned short* p, int sel) {
  V16U t;
  const uint4* q = (const uint4*)(p + sel * 16);
  t.q[0] = q[0];
  t.q[1] = q[1];
  return t.v;
}

__device__ __forceinline__ v8f wmma_bf16(v16bf a, v16bf b, v8f c) {
  return __builtin_amdgcn_wmma_f32_16x16x32_bf16(
      /*neg_a=*/false, a, /*neg_b=*/false, b,
      /*c_mod=*/(short)0, c, /*reuse_a=*/false, /*reuse_b=*/false);
}

__device__ __forceinline__ unsigned short f2bf(float x) {
  unsigned int u = __float_as_uint(x);
  u += 0x7FFFu + ((u >> 16) & 1u);   // round-to-nearest-even
  return (unsigned short)(u >> 16);
}

__device__ __forceinline__ float sigm(float x) { return 1.0f / (1.0f + __expf(-x)); }

// ---------------------------------------------------------------------------
// Prep: f32 -> bf16 weight repack.
// ---------------------------------------------------------------------------
__global__ __launch_bounds__(256) void prep_kernel(
    const float* __restrict__ Wih, const float* __restrict__ Whh,
    const float* __restrict__ bih, const float* __restrict__ bhh,
    const float* __restrict__ Wout,
    unsigned short* __restrict__ WhhB, unsigned short* __restrict__ WihP,
    unsigned short* __restrict__ WoutP, float* __restrict__ bsum) {
  int i = blockIdx.x * 256 + threadIdx.x;
  if (i < 1024 * 256) WhhB[i] = f2bf(Whh[i]);
  if (i < 1024 * 32) {
    int n = i >> 5, k = i & 31;
    WihP[i] = (k < 12) ? f2bf(Wih[n * 12 + k]) : (unsigned short)0;
  }
  if (i < 16 * 256) {
    int n = i >> 8, k = i & 255;
    WoutP[i] = (n < 12) ? f2bf(Wout[n * 256 + k]) : (unsigned short)0;
  }
  if (i < 1024) bsum[i] = bih[i] + bhh[i];
}

// ---------------------------------------------------------------------------
// Context encoder (small: ~1 GFLOP, plain VALU).
// ---------------------------------------------------------------------------
__global__ __launch_bounds__(256) void ctx_kernel(
    const float* __restrict__ obs, const float* __restrict__ act,
    const float* __restrict__ Wctx, const float* __restrict__ bctx,
    const float* __restrict__ Wh0, const float* __restrict__ bh0,
    const float* __restrict__ Wc0, const float* __restrict__ bc0,
    unsigned short* __restrict__ h0ws, float* __restrict__ c0ws) {
  __shared__ float ctx[64 * 128];
  const int tid = threadIdx.x;
  const int rb = blockIdx.x * 64;
  for (int i = tid; i < 64 * 128; i += 256) {
    int r = i >> 7, c = i & 127;
    const float* o = obs + (size_t)(rb + r) * 16;
    const float* a = act + (size_t)(rb + r) * 2;
    const float* w = Wctx + c * 18;
    float s = bctx[c];
#pragma unroll
    for (int q = 0; q < 16; ++q) s += o[q] * w[q];
    s += a[0] * w[16] + a[1] * w[17];
    ctx[i] = s > 0.0f ? s : 0.0f;
  }
  __syncthreads();
  for (int i = tid; i < 64 * 256; i += 256) {
    int r = i >> 8, c = i & 255;
    const float* cx = ctx + r * 128;
    const float* wh = Wh0 + c * 128;
    const float* wc = Wc0 + c * 128;
    float sh = bh0[c], sc = bc0[c];
    for (int k = 0; k < 128; ++k) { sh += cx[k] * wh[k]; sc += cx[k] * wc[k]; }
    size_t gi = (size_t)(rb + r) * 256 + c;
    h0ws[gi] = f2bf(tanhf(sh));
    c0ws[gi] = tanhf(sc);
  }
}

// ---------------------------------------------------------------------------
// Main recurrent kernel.  512 threads = 16 wave32, BM=64 batch rows per block
// held persistently across all 32 steps.  Wave w owns m-tile (w&3) and four
// j-tiles ((w>>2)*4 ..), computing all four gates of each patch so the cell
// update is elementwise on the WMMA C/D layout.
//   hbuf : double-buffered h (bf16)        2 x 32 KB
//   xbuf : feedback input, K-padded (bf16)     4 KB
//   cbuf : cell state in accumulator order    64 KB  (per-wave private slabs)
// ---------------------------------------------------------------------------
#define BM  64
#define JPW 4

__global__ __launch_bounds__(512) void lstm_main_kernel(
    const float* __restrict__ bout, const float* __restrict__ start_tok,
    const unsigned short* __restrict__ WhhB, const unsigned short* __restrict__ WihP,
    const unsigned short* __restrict__ WoutP, const float* __restrict__ bsum,
    const unsigned short* __restrict__ h0ws, const float* __restrict__ c0ws,
    float* __restrict__ out) {
  __shared__ __align__(16) unsigned short hbuf[2][BM * 256];
  __shared__ __align__(16) unsigned short xbuf[BM * 32];
  __shared__ __align__(16) float cbuf[16 * JPW * 32 * 8];

  const int tid  = threadIdx.x;
  const int lane = tid & 31;
  const int wv   = tid >> 5;          // 0..15
  const int sel  = lane >> 4;         // half-wave select for WMMA layouts
  const int ln   = lane & 15;
  const int mt   = wv & 3;            // m-tile (0..3)
  const int jg   = wv >> 2;           // j-group (0..3)
  const int rowbase = blockIdx.x * BM;

  // ---- init: h0 -> LDS, start token -> x, c0 -> cbuf -------------------
  for (int i = tid; i < BM * 256; i += 512)
    hbuf[0][i] = h0ws[(size_t)rowbase * 256 + i];
  for (int i = tid; i < BM * 32; i += 512) {
    int col = i & 31;
    xbuf[i] = (col < 12) ? f2bf(start_tok[col]) : (unsigned short)0;
  }
#pragma unroll
  for (int jj = 0; jj < JPW; ++jj) {
    int j = jg * JPW + jj;
    float* cp = cbuf + (size_t)((wv * JPW + jj) * 32 + lane) * 8;
#pragma unroll
    for (int r = 0; r < 8; ++r) {
      int row = rowbase + mt * 16 + r + sel * 8;
      cp[r] = c0ws[(size_t)row * 256 + j * 16 + ln];
    }
  }

  for (int t = 0; t < 32; ++t) {
    __syncthreads();  // h(t) (in hbuf[t&1]) and x(t) visible

    const unsigned short* hcur = hbuf[t & 1];
    unsigned short*       hnxt = hbuf[(t + 1) & 1];
    const unsigned short* arow = hcur + (mt * 16 + ln) * 256;
    const v16bf ax = load_a16(xbuf + (mt * 16 + ln) * 32, sel);

    // ---- gates + cell update (rolled loops; bounded VGPR pressure) -----
#pragma unroll 1
    for (int jj = 0; jj < JPW; ++jj) {
      const int j = jg * JPW + jj;
      if (jj + 1 < JPW)
        __builtin_prefetch(WhhB + (size_t)((j + 1) * 16 + ln) * 256, 0, 1);

      const unsigned short* bbase = WhhB + (size_t)(j * 16 + ln) * 256;

      v8f acc[4];
#pragma unroll
      for (int g = 0; g < 4; ++g) {
        float bv = bsum[g * 256 + j * 16 + ln];
#pragma unroll
        for (int r = 0; r < 8; ++r) acc[g][r] = bv;
      }
      // x @ Wih^T (K padded to 32): one WMMA per gate
#pragma unroll
      for (int g = 0; g < 4; ++g) {
        v16bf bx = load_b16(WihP + (size_t)(g * 256 + j * 16 + ln) * 32, sel);
        acc[g] = wmma_bf16(ax, bx, acc[g]);
      }
      // h @ Whh^T: K=256 in steps of 32.  Manual ping-pong double buffer:
      // each half consumes its own B set and refills it for the
      // iteration after next -> no register copies, branch-free body.
      // (Refill at k=6 reads <=512B past the j-row: still inside workspace,
      //  value never consumed.)
      v16bf b0[4], b1[4];
#pragma unroll
      for (int g = 0; g < 4; ++g)
        b0[g] = load_b16(bbase + (size_t)g * 65536, sel);
#pragma unroll
      for (int g = 0; g < 4; ++g)
        b1[g] = load_b16(bbase + (size_t)g * 65536 + 32, sel);
#pragma unroll 1
      for (int k = 0; k < 8; k += 2) {
        v16bf a0 = load_a16(arow + k * 32, sel);
#pragma unroll
        for (int g = 0; g < 4; ++g) acc[g] = wmma_bf16(a0, b0[g], acc[g]);
#pragma unroll
        for (int g = 0; g < 4; ++g)
          b0[g] = load_b16(bbase + (size_t)g * 65536 + (k + 2) * 32, sel);
        SCHED_FENCE();
        v16bf a1 = load_a16(arow + (k + 1) * 32, sel);
#pragma unroll
        for (int g = 0; g < 4; ++g) acc[g] = wmma_bf16(a1, b1[g], acc[g]);
#pragma unroll
        for (int g = 0; g < 4; ++g)
          b1[g] = load_b16(bbase + (size_t)g * 65536 + (k + 3) * 32, sel);
        SCHED_FENCE();
      }

      // LSTM cell (torch gate order i,f,g,o) on accumulator layout
      float* cp = cbuf + (size_t)((wv * JPW + jj) * 32 + lane) * 8;
      v8f hv;
#pragma unroll
      for (int r = 0; r < 8; ++r) {
        float iv = sigm(acc[0][r]);
        float fv = sigm(acc[1][r]);
        float gv = tanhf(acc[2][r]);
        float ov = sigm(acc[3][r]);
        float cn = fv * cp[r] + iv * gv;
        cp[r] = cn;
        hv[r] = ov * tanhf(cn);
      }
#pragma unroll
      for (int r = 0; r < 8; ++r) {
        int lrow = mt * 16 + r + sel * 8;
        hnxt[lrow * 256 + j * 16 + ln] = f2bf(hv[r]);
      }
    }

    __syncthreads();  // h(t+1) complete in hbuf[(t+1)&1]

    // ---- pred = h @ Wout^T + bout (N padded to 16); feeds back as x ----
    if (wv < 4) {
      const int m = wv;
      const unsigned short* hrow = hnxt + (m * 16 + ln) * 256;
      float bv = (ln < 12) ? bout[ln] : 0.0f;
      v8f acc;
#pragma unroll
      for (int r = 0; r < 8; ++r) acc[r] = bv;
#pragma unroll
      for (int k = 0; k < 8; ++k) {
        v16bf ah = load_a16(hrow + k * 32, sel);
        v16bf bw = load_b16(WoutP + ln * 256 + k * 32, sel);
        acc = wmma_bf16(ah, bw, acc);
      }
      if (ln < 12) {
#pragma unroll
        for (int r = 0; r < 8; ++r) {
          int lrow = m * 16 + r + sel * 8;
          out[((size_t)(rowbase + lrow) * 32 + t) * 12 + ln] = acc[r];
          xbuf[lrow * 32 + ln] = f2bf(acc[r]);
        }
      }
    }
  }
}

// ---------------------------------------------------------------------------
extern "C" void kernel_launch(void* const* d_in, const int* in_sizes, int n_in,
                              void* d_out, int out_size, void* d_ws, size_t ws_size,
                              hipStream_t stream) {
  (void)in_sizes; (void)n_in; (void)out_size; (void)ws_size;
  const float* obs   = (const float*)d_in[0];
  const float* act   = (const float*)d_in[1];
  const float* Wctx  = (const float*)d_in[2];
  const float* bctx  = (const float*)d_in[3];
  const float* Wh0   = (const float*)d_in[4];
  const float* bh0   = (const float*)d_in[5];
  const float* Wc0   = (const float*)d_in[6];
  const float* bc0   = (const float*)d_in[7];
  const float* Wih   = (const float*)d_in[8];
  const float* Whh   = (const float*)d_in[9];
  const float* bih   = (const float*)d_in[10];
  const float* bhh   = (const float*)d_in[11];
  const float* Wout  = (const float*)d_in[12];
  const float* bout  = (const float*)d_in[13];
  const float* stok  = (const float*)d_in[14];
  float* out = (float*)d_out;

  char* ws = (char*)d_ws;
  size_t off = 0;
  unsigned short* WhhB  = (unsigned short*)(ws + off); off += (size_t)1024 * 256 * 2;
  unsigned short* WihP  = (unsigned short*)(ws + off); off += (size_t)1024 * 32 * 2;
  unsigned short* WoutP = (unsigned short*)(ws + off); off += (size_t)16 * 256 * 2;
  float*          bsum  = (float*)(ws + off);          off += (size_t)1024 * 4;
  unsigned short* h0ws  = (unsigned short*)(ws + off); off += (size_t)8192 * 256 * 2;
  float*          c0ws  = (float*)(ws + off);          off += (size_t)8192 * 256 * 4;

  prep_kernel<<<1024, 256, 0, stream>>>(Wih, Whh, bih, bhh, Wout,
                                        WhhB, WihP, WoutP, bsum);
  ctx_kernel<<<128, 256, 0, stream>>>(obs, act, Wctx, bctx, Wh0, bh0, Wc0, bc0,
                                      h0ws, c0ws);
  lstm_main_kernel<<<128, 512, 0, stream>>>(bout, stok, WhhB, WihP, WoutP, bsum,
                                            h0ws, c0ws, out);
}
